// Detection_Layer_1219770712127
// MI455X (gfx1250) — compile-verified, matrix operand order checked by reading, not run on hardware
//
#include <hip/hip_runtime.h>
#include <hip/hip_bf16.h>

typedef __attribute__((ext_vector_type(16))) __bf16 v16bf;
typedef __attribute__((ext_vector_type(8)))  __bf16 v8bf;
typedef __attribute__((ext_vector_type(8)))  float  v8f;
typedef __attribute__((ext_vector_type(4)))  unsigned int u32x4;
typedef __attribute__((ext_vector_type(8)))  int i32x8;
typedef __attribute__((ext_vector_type(4)))  int i32x4;

#define GRIDW  19
#define GG     361          // 19*19
#define CIN    512
#define CMID   1024
#define NBATCH 64
#define NOUT   255
#define NPAD   256
#define MT     64           // M tile per workgroup
#define NMT    6            // ceil(361/64)

#define ND (NOUT * CMID)        // detect_w elements
#define NX (NBATCH * CIN * GG)  // x elements
#define NW (CMID * CIN * 9)     // conv_w elements

// ---- WMMA fragment load from LDS, wave32 striping (CDNA5 ISA):
// lane<16 holds K 0..7 (V0-3) and 16..23 (V4-7); lane>=16 holds K 8..15 / 24..31.
__device__ __forceinline__ v16bf frag_ld_off(const __bf16* base, int elemOff, int hf) {
    const __bf16* q = base + elemOff + hf * 8;
    union { v16bf v; v8bf h[2]; } u;
    u.h[0] = *(const v8bf*)(q);
    u.h[1] = *(const v8bf*)(q + 16);
    return u.v;
}
__device__ __forceinline__ v16bf frag_ld(const __bf16* base, int row, int hf) {
    return frag_ld_off(base, row * 32, hf);
}

__device__ __forceinline__ float sigf(float x) {
    return 1.0f / (1.0f + __expf(-x));
}

// ---- Tensor Data Mover: 2D bf16 tile -> LDS (CDNA5 ISA).
// data_size=2B, tensor_dim0 = tile_d0 (no column OOB), tensor_dim1 = tensD1
// (rows >= tensD1 are hardware zero-filled in LDS).
__device__ __forceinline__ void tdm_load_2d(void* ldsPtr, const void* gptr,
                                            unsigned tileD0, unsigned tileD1,
                                            unsigned tensD1, unsigned strideElems) {
    unsigned lds = (unsigned)(size_t)ldsPtr;           // low 32 bits = LDS offset
    unsigned long long ga = (unsigned long long)(size_t)gptr;
    u32x4 g0 = { 1u,                                   // count=1, user descriptor
                 lds,                                  // lds_addr
                 (unsigned)ga,                         // global_addr[31:0]
                 ((unsigned)(ga >> 32) & 0x01FFFFFFu) | 0x80000000u };  // addr[56:32] | type=2
    i32x8 g1 = { (int)(1u << 16),                      // data_size=1 -> 2 bytes
                 (int)((tileD0 & 0xFFFFu) << 16),      // tensor_dim0[15:0]
                 (int)(((tileD0 >> 16) & 0xFFFFu) | ((tensD1 & 0xFFFFu) << 16)),
                 (int)(((tensD1 >> 16) & 0xFFFFu) | ((tileD0 & 0xFFFFu) << 16)), // tile_dim0
                 (int)(tileD1 & 0xFFFFu),              // tile_dim1 (tile_dim2=0)
                 (int)strideElems,                     // tensor_dim0_stride[31:0]
                 0, 0 };
    i32x4 z4 = { 0, 0, 0, 0 };
    i32x8 z8 = { 0, 0, 0, 0, 0, 0, 0, 0 };
    __builtin_amdgcn_tensor_load_to_lds(g0, g1, z4, z4, z8, 0);
}

// ---------------- kernel 0: one-shot f32 -> bf16 conversions ----------------
// dwb = detect_w; xb = x (same layout); w2[tap][co][ci] = conv_w[co][ci][tap]
__global__ __launch_bounds__(256)
void cvt_kernel(const float* __restrict__ dw, const float* __restrict__ x,
                const float* __restrict__ cw, __bf16* __restrict__ dwb,
                __bf16* __restrict__ xb, __bf16* __restrict__ w2) {
    int i = blockIdx.x * 256 + threadIdx.x;
    if (i < ND) {
        dwb[i] = (__bf16)dw[i];
    } else if (i < ND + NX) {
        int j = i - ND;
        xb[j] = (__bf16)x[j];
    } else if (i < ND + NX + NW) {
        int k   = i - ND - NX;
        int tap = k >> 19;            // / (CMID*CIN)
        int r   = k & ((CMID * CIN) - 1);
        int co  = r >> 9;             // / CIN
        int ci  = r & (CIN - 1);
        w2[k] = (__bf16)cw[(co * CIN + ci) * 9 + tap];
    }
}

// ---------------- kernel 1: conv 3x3 implicit GEMM (bf16 WMMA + TDM weights) ----
// grid: NBATCH * NMT * (CMID/128); WG tile 64M x 128N, 8 waves (4M x 2N), 4 acc/wave
__global__ __launch_bounds__(256)
void conv1_kernel(const __bf16* __restrict__ xb, const __bf16* __restrict__ w2,
                  const float* __restrict__ cb, __bf16* __restrict__ h) {
    __shared__ __bf16 slab[7 * 21 * 32];   // [ry][xx+1][ci] halo slab, 9408B
    __shared__ __bf16 Bs[128 * 32];        // [co][ci] weight tile, 8KB

    const int bid = blockIdx.x;
    const int n   = bid / (NMT * 8);
    const int rem = bid % (NMT * 8);
    const int m0  = (rem / 8) * MT;
    const int n0  = (rem % 8) * 128;
    const int y0  = m0 / GRIDW;

    const int t    = threadIdx.x;
    const int wave = t >> 5;
    const int lane = t & 31;
    const int wm   = wave >> 1;    // 0..3
    const int wn   = wave & 1;     // 0..1 (64 cols each)
    const int lr   = lane & 15;
    const int hf   = lane >> 4;

    // per-lane output row -> spatial coords (clamped; invalid rows masked at store)
    const int mlane = m0 + wm * 16 + lr;
    const int me    = mlane > 360 ? 360 : mlane;
    const int ly    = me / GRIDW;
    const int lx    = me % GRIDW;

    v8f acc[4] = {};

    const unsigned short* xs = (const unsigned short*)xb;

    for (int kc = 0; kc < CIN; kc += 32) {
        __syncthreads();   // previous iteration's LDS reads complete
        // fill halo slab: rows y0-1 .. y0+5, cols -1..19, 32 ci (packed pairs)
        #pragma unroll
        for (int e = 0; e < 10; ++e) {
            int i = t + 256 * e;
            if (i < 7 * 21 * 16) {
                int ci2 = i & 15;          // ci pair index
                int sp  = i >> 4;          // 0..146
                int xi  = sp % 21;
                int ry  = sp / 21;
                int yy  = y0 - 1 + ry;
                int gx  = xi - 1;
                unsigned val = 0;
                if ((unsigned)yy < GRIDW && (unsigned)gx < GRIDW) {
                    const unsigned short* p =
                        xs + (size_t)(n * CIN + kc + ci2 * 2) * GG + yy * GRIDW + gx;
                    unsigned lo = p[0];
                    unsigned hi = p[GG];
                    val = lo | (hi << 16);
                }
                ((unsigned*)slab)[sp * 16 + ci2] = val;
            }
        }
        for (int tap = 0; tap < 9; ++tap) {
            const int ky = tap / 3 - 1;
            const int kx = tap % 3 - 1;
            __syncthreads();   // slab filled / previous tap's Bs reads done
            if (wave == 0) {   // TDM: async 128x32 bf16 weight tile -> LDS
                tdm_load_2d(Bs, w2 + ((size_t)(tap * CMID + n0)) * CIN + kc,
                            32, 128, 128, CIN);
                __builtin_amdgcn_s_wait_tensorcnt(0);
            }
            __syncthreads();

            int offA = ((ly + ky - y0 + 1) * 21 + (lx + kx + 1)) * 32;
            v16bf a = frag_ld_off(slab, offA, hf);
            #pragma unroll
            for (int j = 0; j < 4; ++j) {
                v16bf b = frag_ld(Bs, wn * 64 + j * 16 + lr, hf);
                acc[j] = __builtin_amdgcn_wmma_f32_16x16x32_bf16(
                    false, a, false, b, (short)0, acc[j], false, false);
            }
        }
    }

    // epilogue: +bias, leaky ReLU, store bf16 h[(n*361+m)][co]
    float bv[4];
    #pragma unroll
    for (int j = 0; j < 4; ++j) bv[j] = cb[n0 + wn * 64 + j * 16 + lr];
    #pragma unroll
    for (int vv = 0; vv < 8; ++vv) {
        int m = m0 + wm * 16 + vv + hf * 8;   // C/D layout: row = vgpr + 8*half
        if (m < GG) {
            size_t rowb = (size_t)(n * GG + m) * CMID;
            #pragma unroll
            for (int j = 0; j < 4; ++j) {
                float f = acc[j][vv] + bv[j];
                f = f > 0.0f ? f : 0.1f * f;
                h[rowb + n0 + wn * 64 + j * 16 + lr] = (__bf16)f;
            }
        }
    }
}

// ---------------- kernel 2: 1x1 conv GEMM (bf16 WMMA + TDM tiles) + head ----------------
// grid: NBATCH * NMT blocks of 256 threads
__global__ __launch_bounds__(256)
void detect_kernel(const __bf16* __restrict__ h, const __bf16* __restrict__ dwb,
                   const float* __restrict__ db, const float* __restrict__ anchors,
                   float* __restrict__ out) {
    // 64KB LDS, overlaid: [As 4KB | Bs 16KB] during GEMM, Of 64KB after.
    __shared__ __align__(16) unsigned char smem[MT * NPAD * 4];
    __bf16* As = (__bf16*)smem;              // [64][32]
    __bf16* Bs = (__bf16*)(smem + 4096);     // [256][32]
    float*  Of = (float*)smem;               // [64][256]

    const int bid = blockIdx.x;
    const int n   = bid / NMT;
    const int m0  = (bid % NMT) * MT;

    const int t    = threadIdx.x;
    const int wave = t >> 5;
    const int lane = t & 31;
    const int wm   = wave & 3;        // M sub-tile 0..3
    const int wn   = wave >> 2;       // N half 0..1 (128 cols each)
    const int lr   = lane & 15;
    const int hf   = lane >> 4;

    v8f acc[8] = {};

    for (int k0 = 0; k0 < CMID; k0 += 32) {
        __syncthreads();   // previous iteration's LDS reads complete
        if (wave == 0) {
            // A: 64x32 bf16 tile of h; rows past 361 hardware zero-filled
            tdm_load_2d(As, h + (size_t)(n * GG + m0) * CMID + k0,
                        32, 64, (unsigned)(GG - m0), CMID);
            // B: 256x32 bf16 tile of detect_w; rows >= 255 hardware zero-filled
            tdm_load_2d(Bs, dwb + k0, 32, 256, NOUT, CMID);
            __builtin_amdgcn_s_wait_tensorcnt(0);
        }
        __syncthreads();

        v16bf a = frag_ld(As, wm * 16 + lr, hf);
        #pragma unroll
        for (int j = 0; j < 8; ++j) {
            v16bf b = frag_ld(Bs, wn * 128 + j * 16 + lr, hf);
            acc[j] = __builtin_amdgcn_wmma_f32_16x16x32_bf16(
                false, a, false, b, (short)0, acc[j], false, false);
        }
    }

    __syncthreads();   // done reading As/Bs -> safe to overlay Of
    #pragma unroll
    for (int j = 0; j < 8; ++j) {
        #pragma unroll
        for (int vv = 0; vv < 8; ++vv) {
            int r = wm * 16 + vv + hf * 8;
            int c = wn * 128 + j * 16 + lr;
            Of[r * NPAD + c] = acc[j][vv];
        }
    }
    __syncthreads();

    // head: one thread per (m, anchor)
    if (t < MT * 3) {
        int mL = t / 3;
        int a  = t % 3;
        int m  = m0 + mL;
        if (m < GG) {
            const float* o  = &Of[mL * NPAD + a * 85];
            const float* bb = &db[a * 85];
            float obj = sigf(o[0] + bb[0]);
            float tx  = sigf(o[1] + bb[1]);
            float ty  = sigf(o[2] + bb[2]);
            float tw  = sigf(o[3] + bb[3]);
            float th  = sigf(o[4] + bb[4]);
            float best = -1.0f;
            int   bi   = 0;
            for (int j = 0; j < 80; ++j) {
                float s = obj * sigf(o[5 + j] + bb[5 + j]);
                if (s > best) { best = s; bi = j; }
            }
            float gx = (float)(m % GRIDW);
            float gy = (float)(m / GRIDW);
            float xc = (tx + gx) * (1.0f / 19.0f);
            float yc = (ty + gy) * (1.0f / 19.0f);
            float bw = __expf(tw) * anchors[a * 2 + 0];
            float bh = __expf(th) * anchors[a * 2 + 1];
            float* op = out + ((size_t)n * (GG * 3) + (size_t)m * 3 + a) * 6;
            op[0] = best;
            op[1] = xc;
            op[2] = yc;
            op[3] = bw;
            op[4] = bh;
            op[5] = (float)bi;
        }
    }
}

extern "C" void kernel_launch(void* const* d_in, const int* in_sizes, int n_in,
                              void* d_out, int out_size, void* d_ws, size_t ws_size,
                              hipStream_t stream) {
    const float* x       = (const float*)d_in[0];
    const float* cw      = (const float*)d_in[1];
    const float* cb      = (const float*)d_in[2];
    const float* dw      = (const float*)d_in[3];
    const float* db      = (const float*)d_in[4];
    const float* anchors = (const float*)d_in[5];
    float* out = (float*)d_out;

    // workspace (bf16 elements): h | dwb | xb | w2  (~81 MB total)
    __bf16* hbuf = (__bf16*)d_ws;
    __bf16* dwb  = hbuf + (size_t)NBATCH * GG * CMID;
    __bf16* xb   = dwb + ND;
    __bf16* w2   = xb + NX;

    const int tot = ND + NX + NW;
    cvt_kernel<<<(tot + 255) / 256, 256, 0, stream>>>(dw, x, cw, dwb, xb, w2);
    conv1_kernel<<<NBATCH * NMT * (CMID / 128), 256, 0, stream>>>(xb, w2, cb, hbuf);
    detect_kernel<<<NBATCH * NMT, 256, 0, stream>>>(hbuf, dwb, db, anchors, out);
}
